// AttentionMM_42631845380745
// MI455X (gfx1250) — matrix operand order repaired and checked by reading, not compiled, not run on hardware
//
#include <hip/hip_runtime.h>
#include <hip/hip_bf16.h>

// ---------------------------------------------------------------------------
// Problem dims (fixed by the reference)
// ---------------------------------------------------------------------------
#define B_DIM 32
#define T_DIM 1024
#define E_DIM 512

#define USE_ASYNC_LDS 1

typedef __bf16 bf16_t;
typedef __attribute__((ext_vector_type(16))) __bf16 v16bf;
typedef __attribute__((ext_vector_type(8)))  __bf16 v8bf;
typedef __attribute__((ext_vector_type(8)))  float  v8f;

// fp32 -> bf16 round-to-nearest-even
__device__ __forceinline__ bf16_t f2bf(float f) {
  union { float f; unsigned u; } v; v.f = f;
  unsigned r = v.u + 0x7fffu + ((v.u >> 16) & 1u);
  unsigned short h = (unsigned short)(r >> 16);
  bf16_t out;
  __builtin_memcpy(&out, &h, 2);
  return out;
}

// ---------------------------------------------------------------------------
// Async global->LDS staging (CDNA5 GLOBAL_LOAD_ASYNC_TO_LDS_B128, ASYNCcnt).
// LDS generic pointers on gfx1250 carry the LDS byte offset in addr[31:0].
// ---------------------------------------------------------------------------
__device__ __forceinline__ void async_copy_b128(void* lds_dst, const void* gsrc) {
#if USE_ASYNC_LDS
  unsigned loff = (unsigned)(uintptr_t)lds_dst;
  unsigned long long g = (unsigned long long)(uintptr_t)gsrc;
  asm volatile("global_load_async_to_lds_b128 %0, %1, off"
               :: "v"(loff), "v"(g) : "memory");
#else
  *(v8bf*)lds_dst = *(const v8bf*)gsrc;
#endif
}

__device__ __forceinline__ void async_wait() {
#if USE_ASYNC_LDS
  asm volatile("s_wait_asynccnt 0x0" ::: "memory");
#endif
}

// ---------------------------------------------------------------------------
// WMMA fragment helpers (wave32, v_wmma_f32_16x16x32_bf16)
// A 16x32 (MxK): lane&15 = row; lanes 0-15: e0..15 = K0..15;
//                lanes 16-31: e0..15 = K16..31.
// B 32x16 (KxN): lane&15 = col; lanes 0-15: e0..15 = K0..15;
//                lanes 16-31: e0..15 = K16..31.
// C 16x16 f32:   VGPR r -> row (lane>>4)*8 + r, col lane&15.
// ---------------------------------------------------------------------------
__device__ __forceinline__ v16bf load_a_frag(const bf16_t* base, int ldk, int lane) {
  const int row = lane & 15;
  const int h = (lane >> 4) * 16;
  const bf16_t* p = base + row * ldk;
  v16bf a;
#pragma unroll
  for (int j = 0; j < 8; ++j) {
    a[j]     = p[h + j];
    a[8 + j] = p[h + 8 + j];
  }
  return a;
}

// baseT points at an LDS tile stored as [n][32] (K-contiguous per column)
__device__ __forceinline__ v16bf load_b_frag(const bf16_t* baseT, int lane) {
  const bf16_t* p = baseT + (lane & 15) * 32 + (lane >> 4) * 16;
  v16bf b;
#pragma unroll
  for (int j = 0; j < 16; ++j) b[j] = p[j];
  return b;
}

__device__ __forceinline__ v8f wmma_bf16(v16bf a, v16bf b, v8f c) {
  return __builtin_amdgcn_wmma_f32_16x16x32_bf16(
      /*neg_a=*/false, a, /*neg_b=*/false, b,
      /*c_mod=*/(short)0, c, /*reuse_a=*/false, /*reuse_b=*/false);
}

// ---------------------------------------------------------------------------
// Kernel: x fp32 [B,T,E] -> bf16 row-major copy AND bf16 transposed [B,E,T]
// ---------------------------------------------------------------------------
__global__ __launch_bounds__(256) void cvt_x_kernel(
    const float* __restrict__ in, bf16_t* __restrict__ outr,
    bf16_t* __restrict__ outt) {
  int i = blockIdx.x * 256 + threadIdx.x;  // grid covers exactly B*T*E
  bf16_t v = f2bf(in[i]);
  outr[i] = v;
  int c = i & (E_DIM - 1);
  int t = (i >> 9) & (T_DIM - 1);   // E = 2^9
  int b = i >> 19;                  // T*E = 2^19
  outt[((size_t)b * E_DIM + c) * T_DIM + t] = v;
}

// ---------------------------------------------------------------------------
// Kernel: W fp32 [E,E] -> bf16 transposed Wt[n][k] = W[k][n]
// ---------------------------------------------------------------------------
__global__ __launch_bounds__(256) void cvt_wT_kernel(
    const float* __restrict__ W, bf16_t* __restrict__ Wt) {
  int i = blockIdx.x * 256 + threadIdx.x;  // grid covers exactly E*E
  int k = i >> 9, n = i & (E_DIM - 1);
  Wt[(size_t)n * E_DIM + k] = f2bf(W[i]);
}

__global__ __launch_bounds__(256) void zero_kernel(float* __restrict__ p, int n) {
  int i = blockIdx.x * 256 + threadIdx.x;
  if (i < n) p[i] = 0.0f;
}

// ---------------------------------------------------------------------------
// Kernel: e = relu(x @ W + bias). x bf16 [B*T,E]; Wt bf16 [E,E] transposed.
// WG = 256 threads (8 waves). Tile 128(M) x 64(N), K-step 32.
// Wave wid: m_sub=(wid&3)*32 (2 m-frags), n_sub=(wid>>2)*32 (2 n-frags);
// 4 WMMAs per K-step per wave.
// ---------------------------------------------------------------------------
__global__ __launch_bounds__(256) void proj_relu_kernel(
    const bf16_t* __restrict__ xbf, const bf16_t* __restrict__ Wt,
    const float* __restrict__ bias, bf16_t* __restrict__ out) {
  __shared__ bf16_t aT[128][32];
  __shared__ bf16_t bT[64][32];
  const int tid = threadIdx.x, lane = tid & 31, wid = tid >> 5;
  const int m_blk = blockIdx.y * 128;
  const int n_blk = blockIdx.x * 64;
  const int m_sub = (wid & 3) * 32;
  const int n_sub = (wid >> 2) * 32;

  v8f acc00 = {}, acc01 = {}, acc10 = {}, acc11 = {};

  for (int k0 = 0; k0 < E_DIM; k0 += 32) {
    __syncthreads();
#pragma unroll
    for (int i = 0; i < 2; ++i) {  // A: 128x32 bf16 = 512 16B chunks
      int c = tid + i * 256;
      int r = c >> 2, col = (c & 3) * 8;
      async_copy_b128(&aT[r][col],
                      xbf + (size_t)(m_blk + r) * E_DIM + k0 + col);
    }
    {  // B: 64x32 bf16 = 256 16B chunks (Wt already [n][k])
      int r = tid >> 2, col = (tid & 3) * 8;
      async_copy_b128(&bT[r][col],
                      Wt + (size_t)(n_blk + r) * E_DIM + k0 + col);
    }
    async_wait();
    __syncthreads();
    v16bf a0 = load_a_frag(&aT[m_sub][0], 32, lane);
    v16bf a1 = load_a_frag(&aT[m_sub + 16][0], 32, lane);
    v16bf b0 = load_b_frag(&bT[n_sub][0], lane);
    v16bf b1 = load_b_frag(&bT[n_sub + 16][0], lane);
    acc00 = wmma_bf16(a0, b0, acc00);
    acc01 = wmma_bf16(a0, b1, acc01);
    acc10 = wmma_bf16(a1, b0, acc10);
    acc11 = wmma_bf16(a1, b1, acc11);
  }

  const int khalf = lane >> 4;
  const int c0 = n_blk + n_sub + (lane & 15), c1 = c0 + 16;
  const int r0 = m_blk + m_sub + khalf * 8, r1 = r0 + 16;
#pragma unroll
  for (int r = 0; r < 8; ++r) {
    int t0 = (r0 + r) & (T_DIM - 1);
    int t1 = (r1 + r) & (T_DIM - 1);
    out[(size_t)(r0 + r) * E_DIM + c0] =
        f2bf(fmaxf(acc00[r] + bias[(size_t)t0 * E_DIM + c0], 0.0f));
    out[(size_t)(r0 + r) * E_DIM + c1] =
        f2bf(fmaxf(acc01[r] + bias[(size_t)t0 * E_DIM + c1], 0.0f));
    out[(size_t)(r1 + r) * E_DIM + c0] =
        f2bf(fmaxf(acc10[r] + bias[(size_t)t1 * E_DIM + c0], 0.0f));
    out[(size_t)(r1 + r) * E_DIM + c1] =
        f2bf(fmaxf(acc11[r] + bias[(size_t)t1 * E_DIM + c1], 0.0f));
  }
}

// ---------------------------------------------------------------------------
// Kernel: scores = e2t @ e1t^T per batch, fused row softmax, bf16 probs out.
// One WG = 16 query rows x all 1024 keys. A rows resident in LDS; B^T frags
// are 16 contiguous bf16 of an e1t row straight from global. Scores spill to
// LDS (aliasing the A area), softmax via wave32 shuffles.
// ---------------------------------------------------------------------------
__global__ __launch_bounds__(256) void attn_softmax_kernel(
    const bf16_t* __restrict__ e2, const bf16_t* __restrict__ e1,
    bf16_t* __restrict__ et) {
  __shared__ union {
    bf16_t arow[16][E_DIM];   // 16 KB (GEMM phase)
    float  s[16][T_DIM];      // 64 KB (softmax phase)
  } sm;
  const int tid = threadIdx.x, lane = tid & 31, wid = tid >> 5;
  const int b = blockIdx.y;
  const int i0 = blockIdx.x * 16;
  const bf16_t* e2base = e2 + ((size_t)b * T_DIM + i0) * E_DIM;

#pragma unroll
  for (int i = 0; i < 4; ++i) {  // 16 rows x 512 = 1024 16B chunks
    int chunk = tid + i * 256;
    int r = chunk >> 6;
    int c = (chunk & 63) * 8;
    async_copy_b128(&sm.arow[r][c], e2base + (size_t)r * E_DIM + c);
  }
  async_wait();
  __syncthreads();

  v8f acc[8];
#pragma unroll
  for (int i = 0; i < 8; ++i) acc[i] = (v8f){};

  const int col16 = lane & 15, khalf = lane >> 4;
  for (int k0 = 0; k0 < E_DIM; k0 += 32) {
    v16bf a = load_a_frag(&sm.arow[0][k0], E_DIM, lane);
#pragma unroll
    for (int nt = 0; nt < 8; ++nt) {
      int n0 = wid * 128 + nt * 16;
      const bf16_t* bp =
          e1 + ((size_t)b * T_DIM + n0 + col16) * E_DIM + k0 + khalf * 16;
      v8bf lo = *(const v8bf*)bp;
      v8bf hi = *(const v8bf*)(bp + 8);
      v16bf bf;
#pragma unroll
      for (int j = 0; j < 8; ++j) { bf[j] = lo[j]; bf[8 + j] = hi[j]; }
      acc[nt] = wmma_bf16(a, bf, acc[nt]);
    }
  }

  __syncthreads();  // arow dead; s aliases it
#pragma unroll
  for (int nt = 0; nt < 8; ++nt) {
    int n = wid * 128 + nt * 16 + col16;
#pragma unroll
    for (int r = 0; r < 8; ++r) sm.s[khalf * 8 + r][n] = acc[nt][r];
  }
  __syncthreads();

#pragma unroll
  for (int rr = 0; rr < 2; ++rr) {
    int row = wid * 2 + rr;
    float mx = -3.4e38f;
#pragma unroll
    for (int i = 0; i < 32; ++i) mx = fmaxf(mx, sm.s[row][lane + i * 32]);
    for (int off = 16; off > 0; off >>= 1) mx = fmaxf(mx, __shfl_xor(mx, off, 32));
    float ex[32];
    float sum = 0.0f;
#pragma unroll
    for (int i = 0; i < 32; ++i) {
      ex[i] = __expf(sm.s[row][lane + i * 32] - mx);
      sum += ex[i];
    }
    for (int off = 16; off > 0; off >>= 1) sum += __shfl_xor(sum, off, 32);
    float inv = 1.0f / sum;
    bf16_t* dst = et + (size_t)b * T_DIM * T_DIM + (size_t)(i0 + row) * T_DIM;
#pragma unroll
    for (int i = 0; i < 32; ++i) dst[lane + i * 32] = f2bf(ex[i] * inv);
  }
}

// ---------------------------------------------------------------------------
// Kernel: batched C = op(et) @ X, bf16 out. B operand comes from Xt [B,E,T]
// (pre-transposed) so its staging is a contiguous async copy.
// TRANS_A=true  -> a1t = et^T @ x2 ; TRANS_A=false -> a2t = et @ x1
// ---------------------------------------------------------------------------
template <bool TRANS_A>
__global__ __launch_bounds__(256) void attn_av_kernel(
    const bf16_t* __restrict__ et, const bf16_t* __restrict__ Xt,
    bf16_t* __restrict__ out) {
  __shared__ bf16_t aT[128][32];
  __shared__ bf16_t bT[64][32];
  const int tid = threadIdx.x, lane = tid & 31, wid = tid >> 5;
  const int b = blockIdx.z;
  const int m_blk = blockIdx.y * 128, n_blk = blockIdx.x * 64;
  const int m_sub = (wid & 3) * 32, n_sub = (wid >> 2) * 32;
  const bf16_t* etb = et + (size_t)b * T_DIM * T_DIM;
  const bf16_t* Xtb = Xt + (size_t)b * E_DIM * T_DIM;

  v8f acc00 = {}, acc01 = {}, acc10 = {}, acc11 = {};

  for (int k0 = 0; k0 < T_DIM; k0 += 32) {
    __syncthreads();
    if (TRANS_A) {  // aT[m][k] = et[k][m] (transpose through LDS)
#pragma unroll
      for (int i = 0; i < 16; ++i) {
        int idx = tid + i * 256;
        int k = idx >> 7, m = idx & 127;
        aT[m][k] = etb[(size_t)(k0 + k) * T_DIM + m_blk + m];
      }
    } else {
#pragma unroll
      for (int i = 0; i < 2; ++i) {
        int c = tid + i * 256;
        int r = c >> 2, col = (c & 3) * 8;
        async_copy_b128(&aT[r][col],
                        etb + (size_t)(m_blk + r) * T_DIM + k0 + col);
      }
    }
    {  // bT[n][k] = X[k][n] = Xt[n][k] : contiguous
      int r = tid >> 2, col = (tid & 3) * 8;
      async_copy_b128(&bT[r][col],
                      Xtb + (size_t)(n_blk + r) * T_DIM + k0 + col);
    }
    async_wait();
    __syncthreads();
    v16bf a0 = load_a_frag(&aT[m_sub][0], 32, lane);
    v16bf a1 = load_a_frag(&aT[m_sub + 16][0], 32, lane);
    v16bf b0 = load_b_frag(&bT[n_sub][0], lane);
    v16bf b1 = load_b_frag(&bT[n_sub + 16][0], lane);
    acc00 = wmma_bf16(a0, b0, acc00);
    acc01 = wmma_bf16(a0, b1, acc01);
    acc10 = wmma_bf16(a1, b0, acc10);
    acc11 = wmma_bf16(a1, b1, acc11);
  }

  const int khalf = lane >> 4;
  const int c0 = n_blk + n_sub + (lane & 15), c1 = c0 + 16;
  const int r0 = m_blk + m_sub + khalf * 8, r1 = r0 + 16;
  bf16_t* ob = out + (size_t)b * T_DIM * E_DIM;
#pragma unroll
  for (int r = 0; r < 8; ++r) {
    ob[(size_t)(r0 + r) * E_DIM + c0] = f2bf(acc00[r]);
    ob[(size_t)(r0 + r) * E_DIM + c1] = f2bf(acc01[r]);
    ob[(size_t)(r1 + r) * E_DIM + c0] = f2bf(acc10[r]);
    ob[(size_t)(r1 + r) * E_DIM + c1] = f2bf(acc11[r]);
  }
}

// ---------------------------------------------------------------------------
// Kernel: o = relu(X@U + A@V + bias); mean over T accumulated into d_out via
// fp32 atomics (scaled 1/T). Two accumulated K-loops into the same WMMA accs.
// Ut/Vt are bf16 pre-transposed [n][k].
// ---------------------------------------------------------------------------
__global__ __launch_bounds__(256) void out_proj_mean_kernel(
    const bf16_t* __restrict__ Xbf, const bf16_t* __restrict__ Abf,
    const bf16_t* __restrict__ Ut, const bf16_t* __restrict__ Vt,
    const float* __restrict__ bias, float* __restrict__ outp, int col_off) {
  __shared__ bf16_t aT[128][32];
  __shared__ bf16_t bT[64][32];
  __shared__ float red[64];
  const int tid = threadIdx.x, lane = tid & 31, wid = tid >> 5;
  const int b = blockIdx.z;
  const int m_blk = blockIdx.y * 128, n_blk = blockIdx.x * 64;
  const int m_sub = (wid & 3) * 32, n_sub = (wid >> 2) * 32;

  v8f acc00 = {}, acc01 = {}, acc10 = {}, acc11 = {};

  for (int ph = 0; ph < 2; ++ph) {
    const bf16_t* Ab = (ph == 0 ? Xbf : Abf) + (size_t)b * T_DIM * E_DIM;
    const bf16_t* Wp = (ph == 0 ? Ut : Vt);
    for (int k0 = 0; k0 < E_DIM; k0 += 32) {
      __syncthreads();
#pragma unroll
      for (int i = 0; i < 2; ++i) {
        int c = tid + i * 256;
        int r = c >> 2, col = (c & 3) * 8;
        async_copy_b128(&aT[r][col],
                        Ab + (size_t)(m_blk + r) * E_DIM + k0 + col);
      }
      {
        int r = tid >> 2, col = (tid & 3) * 8;
        async_copy_b128(&bT[r][col],
                        Wp + (size_t)(n_blk + r) * E_DIM + k0 + col);
      }
      async_wait();
      __syncthreads();
      v16bf a0 = load_a_frag(&aT[m_sub][0], 32, lane);
      v16bf a1 = load_a_frag(&aT[m_sub + 16][0], 32, lane);
      v16bf b0 = load_b_frag(&bT[n_sub][0], lane);
      v16bf b1 = load_b_frag(&bT[n_sub + 16][0], lane);
      acc00 = wmma_bf16(a0, b0, acc00);
      acc01 = wmma_bf16(a0, b1, acc01);
      acc10 = wmma_bf16(a1, b0, acc10);
      acc11 = wmma_bf16(a1, b1, acc11);
    }
  }

  __syncthreads();
  if (tid < 64) red[tid] = 0.0f;
  __syncthreads();

  const int khalf = lane >> 4;
  const int c0l = n_sub + (lane & 15), c1l = c0l + 16;
  const int c0 = n_blk + c0l, c1 = n_blk + c1l;
  const int r0 = m_blk + m_sub + khalf * 8, r1 = r0 + 16;  // == t index
  float ps0 = 0.0f, ps1 = 0.0f;
#pragma unroll
  for (int r = 0; r < 8; ++r) {
    ps0 += fmaxf(acc00[r] + bias[(size_t)(r0 + r) * E_DIM + c0], 0.0f);
    ps1 += fmaxf(acc01[r] + bias[(size_t)(r0 + r) * E_DIM + c1], 0.0f);
    ps0 += fmaxf(acc10[r] + bias[(size_t)(r1 + r) * E_DIM + c0], 0.0f);
    ps1 += fmaxf(acc11[r] + bias[(size_t)(r1 + r) * E_DIM + c1], 0.0f);
  }
  atomicAdd(&red[c0l], ps0);
  atomicAdd(&red[c1l], ps1);
  __syncthreads();
  if (tid < 64) {
    atomicAdd(&outp[(size_t)b * (2 * E_DIM) + col_off + n_blk + tid],
              red[tid] * (1.0f / (float)T_DIM));
  }
}

// ---------------------------------------------------------------------------
// Host launcher. Inputs: x1,x2,W1,W2,U1,U2,V1,V2,b1,b2,b3,b4.
// d_out: fp32 [B, 2E] (concat of o1, o2).
// ---------------------------------------------------------------------------
extern "C" void kernel_launch(void* const* d_in, const int* in_sizes, int n_in,
                              void* d_out, int out_size, void* d_ws,
                              size_t ws_size, hipStream_t stream) {
  const float* x1 = (const float*)d_in[0];
  const float* x2 = (const float*)d_in[1];
  const float* W1 = (const float*)d_in[2];
  const float* W2 = (const float*)d_in[3];
  const float* U1 = (const float*)d_in[4];
  const float* U2 = (const float*)d_in[5];
  const float* V1 = (const float*)d_in[6];
  const float* V2 = (const float*)d_in[7];
  const float* b1 = (const float*)d_in[8];
  const float* b2 = (const float*)d_in[9];
  const float* b3 = (const float*)d_in[10];
  const float* b4 = (const float*)d_in[11];
  float* out = (float*)d_out;

  const size_t NBTE = (size_t)B_DIM * T_DIM * E_DIM;  // 16,777,216
  const size_t NBTT = (size_t)B_DIM * T_DIM * T_DIM;  // 33,554,432
  const size_t NEE  = (size_t)E_DIM * E_DIM;          // 262,144

  char* w = (char*)d_ws;
  bf16_t* x1b  = (bf16_t*)w; w += NBTE * sizeof(bf16_t);
  bf16_t* x2b  = (bf16_t*)w; w += NBTE * sizeof(bf16_t);
  bf16_t* x1bt = (bf16_t*)w; w += NBTE * sizeof(bf16_t);
  bf16_t* x2bt = (bf16_t*)w; w += NBTE * sizeof(bf16_t);
  bf16_t* e1b  = (bf16_t*)w; w += NBTE * sizeof(bf16_t);
  bf16_t* e2b  = (bf16_t*)w; w += NBTE * sizeof(bf16_t);
  bf16_t* a1b  = (bf16_t*)w; w += NBTE * sizeof(bf16_t);
  bf16_t* a2b  = (bf16_t*)w; w += NBTE * sizeof(bf16_t);
  bf16_t* etb  = (bf16_t*)w; w += NBTT * sizeof(bf16_t);
  bf16_t* w1t  = (bf16_t*)w; w += NEE * sizeof(bf16_t);
  bf16_t* w2t  = (bf16_t*)w; w += NEE * sizeof(bf16_t);
  bf16_t* u1t  = (bf16_t*)w; w += NEE * sizeof(bf16_t);
  bf16_t* u2t  = (bf16_t*)w; w += NEE * sizeof(bf16_t);
  bf16_t* v1t  = (bf16_t*)w; w += NEE * sizeof(bf16_t);
  bf16_t* v2t  = (bf16_t*)w; w += NEE * sizeof(bf16_t);
  (void)ws_size; (void)n_in; (void)in_sizes;

  // 1) input conversions (row-major + transposed) and weight conv+transpose
  {
    dim3 g((unsigned)(NBTE / 256));
    cvt_x_kernel<<<g, 256, 0, stream>>>(x1, x1b, x1bt);
    cvt_x_kernel<<<g, 256, 0, stream>>>(x2, x2b, x2bt);
    dim3 gw((unsigned)(NEE / 256));
    cvt_wT_kernel<<<gw, 256, 0, stream>>>(W1, w1t);
    cvt_wT_kernel<<<gw, 256, 0, stream>>>(W2, w2t);
    cvt_wT_kernel<<<gw, 256, 0, stream>>>(U1, u1t);
    cvt_wT_kernel<<<gw, 256, 0, stream>>>(U2, u2t);
    cvt_wT_kernel<<<gw, 256, 0, stream>>>(V1, v1t);
    cvt_wT_kernel<<<gw, 256, 0, stream>>>(V2, v2t);
  }
  // 2) projections + relu
  {
    dim3 g(E_DIM / 64, (B_DIM * T_DIM) / 128);  // (8, 256)
    proj_relu_kernel<<<g, 256, 0, stream>>>(x1b, w1t, b1, e1b);
    proj_relu_kernel<<<g, 256, 0, stream>>>(x2b, w2t, b2, e2b);
  }
  // 3) scores + softmax -> bf16 probs
  {
    dim3 g(T_DIM / 16, B_DIM);  // (64, 32)
    attn_softmax_kernel<<<g, 256, 0, stream>>>(e2b, e1b, etb);
  }
  // 4) attention-value GEMMs
  {
    dim3 g(E_DIM / 64, T_DIM / 128, B_DIM);  // (8, 8, 32)
    attn_av_kernel<true><<<g, 256, 0, stream>>>(etb, x2bt, a1b);   // a1t
    attn_av_kernel<false><<<g, 256, 0, stream>>>(etb, x1bt, a2b);  // a2t
  }
  // 5) zero output, then fused output projections + mean
  {
    dim3 gz((out_size + 255) / 256);
    zero_kernel<<<gz, 256, 0, stream>>>(out, out_size);
    dim3 g(E_DIM / 64, T_DIM / 128, B_DIM);  // (8, 8, 32)
    out_proj_mean_kernel<<<g, 256, 0, stream>>>(x1b, a1b, u1t, v1t, b3, out, 0);
    out_proj_mean_kernel<<<g, 256, 0, stream>>>(x2b, a2b, u2t, v2t, b4, out,
                                                E_DIM);
  }
}
